// AttentionReadout_56556129353758
// MI455X (gfx1250) — compile-verified, matrix operand order checked by reading, not run on hardware
//
#include <hip/hip_runtime.h>
#include <math.h>

// ---------------- types ----------------
typedef __bf16 bf16_t;
typedef bf16_t v16bf __attribute__((ext_vector_type(16)));
typedef float  v8f   __attribute__((ext_vector_type(8)));

// ---------------- tiling ----------------
#define DIM       256
#define TILE_M    128
#define NTHREADS  256
#define FPAD      260   // f32 LDS row stride (banks shift by 4 per row -> conflict free b128)
#define WPAD      264   // bf16 LDS row stride (528B row = 132 dwords -> shift 4 banks per row)

// ---------------- LDS layout (dynamic) ----------------
#define S_FEAT_BYTES (TILE_M * FPAD * 4)          // 133120
#define S_W_OFF      (S_FEAT_BYTES)
#define S_W_BYTES    (DIM * WPAD * 2)             // 135168
#define S_SEG_OFF    (S_W_OFF + S_W_BYTES)        // 268288
#define S_PART_OFF   (S_SEG_OFF + TILE_M * 4)     // 268800
#define S_ALPHA_OFF  (S_PART_OFF + TILE_M * 4)    // 269312
#define S_WE_OFF     (S_ALPHA_OFF + TILE_M * 4)   // 269824
#define SMEM_BYTES   (S_WE_OFF + DIM * 4)         // 270848  (< 320KB WGP budget)

// A fragment: 16x32 bf16, M = lane%16, K = kt*32 + (lane<16 ? {0..7,16..23} : {8..15,24..31})
__device__ __forceinline__ v16bf load_a_frag(const float* s_feat, int row, int kbase) {
    const float* p = s_feat + row * FPAD + kbase;
    const float4 f0 = *(const float4*)(p + 0);
    const float4 f1 = *(const float4*)(p + 4);
    const float4 f2 = *(const float4*)(p + 16);
    const float4 f3 = *(const float4*)(p + 20);
    v16bf a;
    a[0]  = (bf16_t)f0.x; a[1]  = (bf16_t)f0.y; a[2]  = (bf16_t)f0.z; a[3]  = (bf16_t)f0.w;
    a[4]  = (bf16_t)f1.x; a[5]  = (bf16_t)f1.y; a[6]  = (bf16_t)f1.z; a[7]  = (bf16_t)f1.w;
    a[8]  = (bf16_t)f2.x; a[9]  = (bf16_t)f2.y; a[10] = (bf16_t)f2.z; a[11] = (bf16_t)f2.w;
    a[12] = (bf16_t)f3.x; a[13] = (bf16_t)f3.y; a[14] = (bf16_t)f3.z; a[15] = (bf16_t)f3.w;
    return a;
}

// B fragment: B[k][n] = W[n][k] (out = X @ W^T). N = lane%16, same K split as A.
// s_w holds W row-major in bf16, so a fragment is two 16B chunks of row n.
__device__ __forceinline__ v16bf load_b_frag(const bf16_t* s_w, int n, int kbase) {
    const bf16_t* p = s_w + n * WPAD + kbase;
    union { uint4 q[2]; v16bf v; } u;
    u.q[0] = *(const uint4*)(p);
    u.q[1] = *(const uint4*)(p + 16);
    return u.v;
}

// One wave computes a 64x64 tile: rows [wrow,wrow+64), cols [wcol,wcol+64)
__device__ __forceinline__ void wave_gemm(const float* s_feat, const bf16_t* s_w,
                                          int wrow, int wcol, int lane, v8f acc[4][4]) {
    const int nlo  = lane & 15;
    const int koff = (lane & 16) ? 8 : 0;
#pragma unroll
    for (int kt = 0; kt < 8; ++kt) {
        const int kbase = kt * 32 + koff;
        v16bf a[4], b[4];
#pragma unroll
        for (int mt = 0; mt < 4; ++mt) a[mt] = load_a_frag(s_feat, wrow + mt * 16 + nlo, kbase);
#pragma unroll
        for (int nt = 0; nt < 4; ++nt) b[nt] = load_b_frag(s_w, wcol + nt * 16 + nlo, kbase);
#pragma unroll
        for (int mt = 0; mt < 4; ++mt)
#pragma unroll
            for (int nt = 0; nt < 4; ++nt)
                acc[mt][nt] = __builtin_amdgcn_wmma_f32_16x16x32_bf16(
                    false, a[mt], false, b[nt], (short)0, acc[mt][nt], false, false);
    }
}

__device__ __forceinline__ void stage_w_bf16(const float* __restrict__ W, bf16_t* s_w, int tid) {
#pragma unroll 4
    for (int idx = tid; idx < DIM * (DIM / 4); idx += NTHREADS) {
        const int row = idx >> 6;
        const int c4  = (idx & 63) << 2;
        const float4 f = *(const float4*)(W + row * DIM + c4);
        bf16_t* p = s_w + row * WPAD + c4;
        p[0] = (bf16_t)f.x; p[1] = (bf16_t)f.y; p[2] = (bf16_t)f.z; p[3] = (bf16_t)f.w;
    }
}

// ---------------------------------------------------------------------------
// Kernel 1: feat_v[b,:] = feat[last_nodes[b],:] @ W_v^T + b_v   (B x 256)
// ---------------------------------------------------------------------------
__global__ __launch_bounds__(NTHREADS) void attn_featv_kernel(
    const float* __restrict__ feat, const int* __restrict__ last_nodes,
    const float* __restrict__ Wv, const float* __restrict__ bv,
    float* __restrict__ featv, int Btot) {
    extern __shared__ char smem[];
    float*  s_feat = (float*)smem;
    bf16_t* s_w    = (bf16_t*)(smem + S_W_OFF);
    int*    s_idx  = (int*)(smem + S_SEG_OFF);

    const int tid = threadIdx.x;
    const int r0  = blockIdx.x * TILE_M;

    if (tid < TILE_M) s_idx[tid] = (r0 + tid < Btot) ? last_nodes[r0 + tid] : 0;
    __syncthreads();

    for (int idx = tid; idx < TILE_M * (DIM / 4); idx += NTHREADS) {
        const int row = idx >> 6;
        const int c4  = (idx & 63) << 2;
        const int node = s_idx[row];
        const float4 f = *(const float4*)(feat + (size_t)node * DIM + c4);
        *(float4*)(s_feat + row * FPAD + c4) = f;
    }
    stage_w_bf16(Wv, s_w, tid);
    __syncthreads();

    const int wave = tid >> 5, lane = tid & 31;
    const int wrow = (wave >> 2) * 64, wcol = (wave & 3) * 64;
    const int nlo  = lane & 15;
    const int mha  = (lane & 16) ? 8 : 0;

    v8f acc[4][4];
    const v8f z = {0.f, 0.f, 0.f, 0.f, 0.f, 0.f, 0.f, 0.f};
#pragma unroll
    for (int mt = 0; mt < 4; ++mt)
#pragma unroll
        for (int nt = 0; nt < 4; ++nt) acc[mt][nt] = z;

    wave_gemm(s_feat, s_w, wrow, wcol, lane, acc);

#pragma unroll
    for (int mt = 0; mt < 4; ++mt)
#pragma unroll
        for (int nt = 0; nt < 4; ++nt) {
            const int col = wcol + nt * 16 + nlo;
            const float b = bv[col];
#pragma unroll
            for (int r = 0; r < 8; ++r) {
                const int row = r0 + wrow + mt * 16 + r + mha;
                if (row < Btot) featv[(size_t)row * DIM + col] = acc[mt][nt][r] + b;
            }
        }
}

// ---------------------------------------------------------------------------
// Kernel 2 (fused): u = feat@Wu^T ; e = sigmoid(u + feat_v[seg]) . w_e ;
//                   alpha = e*cnt ; out[seg] += feat*alpha  (sorted seg runs)
// ---------------------------------------------------------------------------
__global__ __launch_bounds__(NTHREADS) void attn_main_kernel(
    const float* __restrict__ feat, const float* __restrict__ cnt,
    const int* __restrict__ seg, const float* __restrict__ Wu,
    const float* __restrict__ featv, const float* __restrict__ we,
    float* __restrict__ out, int Ntot) {
    extern __shared__ char smem[];
    float*  s_feat  = (float*)smem;
    bf16_t* s_w     = (bf16_t*)(smem + S_W_OFF);
    int*    s_seg   = (int*)(smem + S_SEG_OFF);
    float*  s_part  = (float*)(smem + S_PART_OFF);
    float*  s_alpha = (float*)(smem + S_ALPHA_OFF);
    float*  s_we    = (float*)(smem + S_WE_OFF);

    const int tid = threadIdx.x;
    const int r0  = blockIdx.x * TILE_M;

    // ---- stage feat tile (f32, single HBM pass), Wu (bf16), seg ids, w_e ----
    for (int idx = tid; idx < TILE_M * (DIM / 4); idx += NTHREADS) {
        const int row = idx >> 6;
        const int c4  = (idx & 63) << 2;
        const int node = r0 + row;
        float4 f = {0.f, 0.f, 0.f, 0.f};
        if (node < Ntot) f = *(const float4*)(feat + (size_t)node * DIM + c4);
        *(float4*)(s_feat + row * FPAD + c4) = f;
    }
    stage_w_bf16(Wu, s_w, tid);
    if (tid < TILE_M) {
        const int node = r0 + tid;
        s_seg[tid]  = seg[node < Ntot ? node : (Ntot - 1)];
        s_part[tid] = 0.f;
    }
    if (tid < DIM) s_we[tid] = we[tid];
    __syncthreads();

    const int wave = tid >> 5, lane = tid & 31;
    const int wrow = (wave >> 2) * 64, wcol = (wave & 3) * 64;
    const int nlo  = lane & 15;
    const int mha  = (lane & 16) ? 8 : 0;

    v8f acc[4][4];
    const v8f z = {0.f, 0.f, 0.f, 0.f, 0.f, 0.f, 0.f, 0.f};
#pragma unroll
    for (int mt = 0; mt < 4; ++mt)
#pragma unroll
        for (int nt = 0; nt < 4; ++nt) acc[mt][nt] = z;

    wave_gemm(s_feat, s_w, wrow, wcol, lane, acc);

    // ---- gate: partial_row = sum_cols sigmoid(u + feat_v[seg][col]) * w_e[col] ----
#pragma unroll
    for (int mt = 0; mt < 4; ++mt) {
#pragma unroll
        for (int r = 0; r < 8; ++r) {
            const int m = wrow + mt * 16 + r + mha;       // tile-local row
            const float* fv = featv + (size_t)s_seg[m] * DIM;
            float partial = 0.f;
#pragma unroll
            for (int nt = 0; nt < 4; ++nt) {
                const int col = wcol + nt * 16 + nlo;
                const float u = acc[mt][nt][r] + fv[col];
                partial += s_we[col] / (1.0f + __expf(-u));
            }
            // reduce over the 16 lanes holding this row's 64 columns
#pragma unroll
            for (int off = 1; off < 16; off <<= 1)
                partial += __shfl_xor(partial, off, 16);
            if (nlo == 0) atomicAdd(&s_part[m], partial); // combine the 4 column-waves
        }
    }
    __syncthreads();

    if (tid < TILE_M) {
        const int node = r0 + tid;
        s_alpha[tid] = (node < Ntot) ? s_part[tid] * cnt[node] : 0.f;
    }
    __syncthreads();

    // ---- segment-sum readout: thread per column, run-length scan over sorted segs ----
    {
        const int c = tid;                 // NTHREADS == DIM
        int rmax = Ntot - r0; if (rmax > TILE_M) rmax = TILE_M;
        int cur = s_seg[0];
        float a = 0.f;
        for (int r = 0; r < rmax; ++r) {
            const int s = s_seg[r];
            if (s != cur) {
                atomicAdd(out + (size_t)cur * DIM + c, a);
                a = 0.f; cur = s;
            }
            a += s_feat[r * FPAD + c] * s_alpha[r];
        }
        atomicAdd(out + (size_t)cur * DIM + c, a);
    }
}

__global__ void zero_f32_kernel(float* __restrict__ p, int n) {
    const int i = blockIdx.x * blockDim.x + threadIdx.x;
    if (i < n) p[i] = 0.f;
}

// ---------------------------------------------------------------------------
extern "C" void kernel_launch(void* const* d_in, const int* in_sizes, int n_in,
                              void* d_out, int out_size, void* d_ws, size_t ws_size,
                              hipStream_t stream) {
    (void)n_in; (void)ws_size;
    const float* feat = (const float*)d_in[0];
    const float* cnt  = (const float*)d_in[1];
    const int*   seg  = (const int*)d_in[2];
    const int*   last = (const int*)d_in[3];
    const float* Wu   = (const float*)d_in[4];
    const float* Wv   = (const float*)d_in[5];
    const float* bv   = (const float*)d_in[6];
    const float* we   = (const float*)d_in[7];
    float* out = (float*)d_out;

    const int Ntot = in_sizes[1];   // cnt: [N]
    const int Btot = in_sizes[3];   // last_nodes: [B]
    float* featv = (float*)d_ws;    // [B, 256] scratch

    // opt into >64KB dynamic LDS (idempotent, capture-safe: no stream ops)
    hipFuncSetAttribute(reinterpret_cast<const void*>(attn_featv_kernel),
                        hipFuncAttributeMaxDynamicSharedMemorySize, SMEM_BYTES);
    hipFuncSetAttribute(reinterpret_cast<const void*>(attn_main_kernel),
                        hipFuncAttributeMaxDynamicSharedMemorySize, SMEM_BYTES);

    zero_f32_kernel<<<(out_size + 255) / 256, 256, 0, stream>>>(out, out_size);
    attn_featv_kernel<<<(Btot + TILE_M - 1) / TILE_M, NTHREADS, SMEM_BYTES, stream>>>(
        feat, last, Wv, bv, featv, Btot);
    attn_main_kernel<<<(Ntot + TILE_M - 1) / TILE_M, NTHREADS, SMEM_BYTES, stream>>>(
        feat, cnt, seg, Wu, featv, we, out, Ntot);
}
